// TexualEmbeddingLayer_42984032698690
// MI455X (gfx1250) — compile-verified, matrix operand order checked by reading, not run on hardware
//
#include <hip/hip_runtime.h>
#include <hip/hip_bf16.h>
#include <math.h>

// ---------------------------------------------------------------------------
// Reference dims: BS=256, S=256, D_IN=512, D_EMB=1024, k=76, H=512.
// Rows padded to 96 per batch so 32-row (2x16) WMMA macro-tiles never
// straddle a batch boundary. Padded rows are masked out of BN stats and pool.
// ---------------------------------------------------------------------------
constexpr int BS    = 256;
constexpr int SEQ   = 256;
constexpr int DIN   = 512;
constexpr int DEMB  = 1024;
constexpr int KSEL  = 76;
constexpr int MPAD  = 96;              // 3 macro-tiles of 32 rows per batch
constexpr int MTOT  = BS * MPAD;       // 24576 padded rows
constexpr int MMACS = MTOT / 32;       // 768 M macro-tiles
constexpr float NROWS = 19456.0f;      // BS*KSEL real rows for BN stats
constexpr float BN_EPS = 1e-5f;

typedef __attribute__((ext_vector_type(16))) __bf16 v16bf;
typedef __attribute__((ext_vector_type(8)))  float  v8f;

__device__ __forceinline__ unsigned short f2bf(float x) {
    unsigned int u = __float_as_uint(x);
    u += 0x7FFFu + ((u >> 16) & 1u);        // round-to-nearest-even
    return (unsigned short)(u >> 16);
}

// Order-preserving float -> uint key (for atomicMax-based pooling).
__device__ __forceinline__ unsigned int fkey(float f) {
    unsigned int b = __float_as_uint(f);
    return (b & 0x80000000u) ? ~b : (b | 0x80000000u);
}
__device__ __forceinline__ float funkey(unsigned int k) {
    return (k & 0x80000000u) ? __uint_as_float(k & 0x7FFFFFFFu)
                             : __uint_as_float(~k);
}

// 16x32 bf16 A-fragment slice for this lane (ISA 7.12.2): lane holds row
// M = lane%16; elements 0..7 = K[8*hf..8*hf+7], 8..15 = K[16+8*hf..23+8*hf]
// -> two contiguous b128 loads per lane.
__device__ __forceinline__ v16bf load_frag(const unsigned short* row, int kk, int hf) {
    union { struct { uint4 lo; uint4 hi; } q; v16bf v; } u;
    u.q.lo = *reinterpret_cast<const uint4*>(row + kk + 8 * hf);
    u.q.hi = *reinterpret_cast<const uint4*>(row + kk + 16 + 8 * hf);
    return u.v;
}

// 32x32 macro-tile, K=512: 2 A-frags + 2 B-frags feed 4 WMMAs per k-step
// (halves per-WMMA fragment traffic vs 16x16 tiles; L2-bandwidth bound).
// A row-major (M,K); B pre-transposed (N,K). Accumulates into c[4]:
// c[0]=(m0,n0) c[1]=(m0,n1) c[2]=(m1,n0) c[3]=(m1,n1).
__device__ __forceinline__ void gemm32x32_k512(const unsigned short* __restrict__ A,
                                               const unsigned short* __restrict__ Bt,
                                               int m0, int n0, int l15, int hf,
                                               v8f c[4]) {
    const unsigned short* A0 = A  + (size_t)(m0 + l15) * DIN;
    const unsigned short* A1 = A0 + (size_t)16 * DIN;
    const unsigned short* B0 = Bt + (size_t)(n0 + l15) * DIN;
    const unsigned short* B1 = B0 + (size_t)16 * DIN;
#pragma unroll 2
    for (int kk = 0; kk < DIN; kk += 32) {
        v16bf a0 = load_frag(A0, kk, hf);
        v16bf a1 = load_frag(A1, kk, hf);
        v16bf b0 = load_frag(B0, kk, hf);
        v16bf b1 = load_frag(B1, kk, hf);
        c[0] = __builtin_amdgcn_wmma_f32_16x16x32_bf16(false, a0, false, b0, (short)0, c[0], false, false);
        c[1] = __builtin_amdgcn_wmma_f32_16x16x32_bf16(false, a0, false, b1, (short)0, c[1], false, false);
        c[2] = __builtin_amdgcn_wmma_f32_16x16x32_bf16(false, a1, false, b0, (short)0, c[2], false, false);
        c[3] = __builtin_amdgcn_wmma_f32_16x16x32_bf16(false, a1, false, b1, (short)0, c[3], false, false);
    }
}

// ---------------------------------------------------------------------------
// K0: per-batch token count -> L, pool_len; gather rows (L+i)&255 of features
// (closed form of the degenerate top_k), L2-normalize, bf16 into Xbf
// (96 rows/batch, rows 76..95 zeroed). One block/batch, one wave/row.
// ---------------------------------------------------------------------------
__global__ void k0_gather(const float* __restrict__ features,
                          const int*   __restrict__ text,
                          unsigned short* __restrict__ Xbf,
                          int* __restrict__ pool_len) {
    const int b    = blockIdx.x;
    const int tid  = threadIdx.x;
    const int lane = tid & 31;
    const int wv   = tid >> 5;

    __shared__ int sL;
    if (tid == 0) sL = 0;
    __syncthreads();
    if (text[b * SEQ + tid] != 0) atomicAdd(&sL, 1);
    __syncthreads();
    const int L = sL;
    if (tid == 0) {
        int p = L - 2;
        p = p < 1 ? 1 : (p > KSEL ? KSEL : p);
        pool_len[b] = p;
    }

    for (int i = wv; i < MPAD; i += 8) {
        unsigned short* dst = Xbf + (size_t)(b * MPAD + i) * DIN + lane * 16;
        if (i < KSEL) {
            const int src = (L + i) & (SEQ - 1);
            const float4* f4 = reinterpret_cast<const float4*>(
                features + ((size_t)b * SEQ + src) * DIN + lane * 16);
            float4 q0 = f4[0], q1 = f4[1], q2 = f4[2], q3 = f4[3];
            float s = q0.x*q0.x + q0.y*q0.y + q0.z*q0.z + q0.w*q0.w
                    + q1.x*q1.x + q1.y*q1.y + q1.z*q1.z + q1.w*q1.w
                    + q2.x*q2.x + q2.y*q2.y + q2.z*q2.z + q2.w*q2.w
                    + q3.x*q3.x + q3.y*q3.y + q3.z*q3.z + q3.w*q3.w;
#pragma unroll
            for (int off = 16; off > 0; off >>= 1) s += __shfl_xor(s, off, 32);
            const float inv = 1.0f / fmaxf(sqrtf(s), 1e-6f);
            const float v[16] = { q0.x,q0.y,q0.z,q0.w, q1.x,q1.y,q1.z,q1.w,
                                  q2.x,q2.y,q2.z,q2.w, q3.x,q3.y,q3.z,q3.w };
#pragma unroll
            for (int j = 0; j < 16; ++j) dst[j] = f2bf(v[j] * inv);
        } else {
#pragma unroll
            for (int j = 0; j < 16; ++j) dst[j] = 0;
        }
    }
}

// ---------------------------------------------------------------------------
// K1: weights -> bf16, transposed to (N,K); fused output bias (lin_b+b2);
// zero BN stat accumulators; init pooled output keys to the minimum key.
// ---------------------------------------------------------------------------
__global__ void k1_prep(const float* __restrict__ lin_w,  // (512,1024)
                        const float* __restrict__ w1,     // (512,512)
                        const float* __restrict__ w2,     // (512,1024)
                        const float* __restrict__ lin_b,
                        const float* __restrict__ b2,
                        unsigned short* __restrict__ LinWt, // (1024,512)
                        unsigned short* __restrict__ W1t,   // (512,512)
                        unsigned short* __restrict__ W2t,   // (1024,512)
                        float* __restrict__ b_out,
                        float* __restrict__ stats,
                        unsigned int* __restrict__ out_u) {
    const int idx = blockIdx.x * blockDim.x + threadIdx.x;  // 0 .. 524287
    const int n = idx >> 9;
    const int k = idx & 511;
    LinWt[idx] = f2bf(lin_w[(size_t)k * DEMB + n]);
    W2t[idx]   = f2bf(w2[(size_t)k * DEMB + n]);
    if (idx < 512 * 512) W1t[idx] = f2bf(w1[(size_t)k * 512 + n]);
    if (idx < BS * DEMB) out_u[idx] = 0u;                 // min possible key
    if (idx < DEMB) b_out[idx] = lin_b[idx] + b2[idx];
    if (idx < 1024) stats[idx] = 0.0f;                    // 512 sums + 512 sumsq
}

// ---------------------------------------------------------------------------
// K2a: GEMM1 stats pass (Y = X @ W1; b1 cancels in train-mode BN).
// 8 waves/block, one 32x32 macro-tile per wave. Row validity mask uses the
// scalar per-block base (mbase) -- no per-lane integer division.
// ---------------------------------------------------------------------------
__global__ void k2_gemm1_stats(const unsigned short* __restrict__ Xbf,
                               const unsigned short* __restrict__ W1t,
                               float* __restrict__ stats) {
    const int lane = threadIdx.x & 31;
    const int wv   = threadIdx.x >> 5;
    const int hf   = lane >> 4;
    const int l15  = lane & 15;
    const int m0    = blockIdx.y * 32;
    const int mbase = (blockIdx.y % 3) * 32;             // row base within batch
    const int n0    = (blockIdx.x * 8 + wv) * 32;

    v8f c[4] = {};
    gemm32x32_k512(Xbf, W1t, m0, n0, l15, hf, c);

    float s[2] = {0.f, 0.f}, sq[2] = {0.f, 0.f};
#pragma unroll
    for (int mi = 0; mi < 2; ++mi)
#pragma unroll
        for (int r = 0; r < 8; ++r) {
            if (mbase + 16 * mi + r + 8 * hf < KSEL) {
                float y0 = c[mi * 2 + 0][r], y1 = c[mi * 2 + 1][r];
                s[0] += y0; sq[0] += y0 * y0;
                s[1] += y1; sq[1] += y1 * y1;
            }
        }
#pragma unroll
    for (int ni = 0; ni < 2; ++ni) {
        s[ni]  += __shfl_xor(s[ni], 16, 32);
        sq[ni] += __shfl_xor(sq[ni], 16, 32);
    }
    if (hf == 0) {
#pragma unroll
        for (int ni = 0; ni < 2; ++ni) {
            atomicAdd(&stats[n0 + 16 * ni + l15], s[ni]);
            atomicAdd(&stats[512 + n0 + 16 * ni + l15], sq[ni]);
        }
    }
}

// K3: fold stats into BN scale/shift.
__global__ void k3_finalize(const float* __restrict__ stats,
                            const float* __restrict__ g1,
                            const float* __restrict__ be1,
                            float* __restrict__ scaleArr,
                            float* __restrict__ shiftArr) {
    const int c = blockIdx.x * blockDim.x + threadIdx.x;
    if (c >= 512) return;
    const float mean = stats[c] / NROWS;
    const float var  = stats[512 + c] / NROWS - mean * mean;
    const float sc   = g1[c] * rsqrtf(var + BN_EPS);
    scaleArr[c] = sc;
    shiftArr[c] = be1[c] - mean * sc;
}

// K2b: GEMM1 apply pass -> H = bf16(relu(Y*scale + shift)).
__global__ void k2_gemm1_apply(const unsigned short* __restrict__ Xbf,
                               const unsigned short* __restrict__ W1t,
                               const float* __restrict__ scaleArr,
                               const float* __restrict__ shiftArr,
                               unsigned short* __restrict__ Hbf) {
    const int lane = threadIdx.x & 31;
    const int wv   = threadIdx.x >> 5;
    const int hf   = lane >> 4;
    const int l15  = lane & 15;
    const int m0   = blockIdx.y * 32;
    const int n0   = (blockIdx.x * 8 + wv) * 32;

    v8f c[4] = {};
    gemm32x32_k512(Xbf, W1t, m0, n0, l15, hf, c);

#pragma unroll
    for (int ni = 0; ni < 2; ++ni) {
        const int col = n0 + 16 * ni + l15;
        const float sc = scaleArr[col], sh = shiftArr[col];
#pragma unroll
        for (int mi = 0; mi < 2; ++mi)
#pragma unroll
            for (int r = 0; r < 8; ++r) {
                const int row = m0 + 16 * mi + r + 8 * hf;
                Hbf[(size_t)row * DIN + col] =
                    f2bf(fmaxf(c[mi * 2 + ni][r] * sc + sh, 0.0f));
            }
    }
}

// ---------------------------------------------------------------------------
// K4: fused O = X@lin_w + H@w2 as one flat 2x2-blocked WMMA GEMM (both
// products share the accumulators). Epilogue: masked in-register row-max
// (macro-tile is batch-local), shfl_xor(16) merges C-fragment halves, bias
// added post-max, one order-preserving-uint atomicMax per output element.
// ---------------------------------------------------------------------------
__global__ void k4_gemm2_pool(const unsigned short* __restrict__ Xbf,
                              const unsigned short* __restrict__ Hbf,
                              const unsigned short* __restrict__ LinWt,
                              const unsigned short* __restrict__ W2t,
                              const float* __restrict__ b_out,
                              const int* __restrict__ pool_len,
                              unsigned int* __restrict__ out_u) {
    const int lane = threadIdx.x & 31;
    const int wv   = threadIdx.x >> 5;
    const int hf   = lane >> 4;
    const int l15  = lane & 15;
    const int m0    = blockIdx.y * 32;
    const int b     = blockIdx.y / 3;                    // batch of this macro-tile
    const int mbase = (blockIdx.y % 3) * 32;
    const int n0    = (blockIdx.x * 8 + wv) * 32;
    const int pool  = pool_len[b];

    v8f c[4] = {};
    gemm32x32_k512(Xbf, LinWt, m0, n0, l15, hf, c);      // linear branch
    gemm32x32_k512(Hbf, W2t,  m0, n0, l15, hf, c);       // MLP branch (summed)

    float mx[2] = {-INFINITY, -INFINITY};
#pragma unroll
    for (int mi = 0; mi < 2; ++mi)
#pragma unroll
        for (int r = 0; r < 8; ++r) {
            if (mbase + 16 * mi + r + 8 * hf < pool) {
                mx[0] = fmaxf(mx[0], c[mi * 2 + 0][r]);
                mx[1] = fmaxf(mx[1], c[mi * 2 + 1][r]);
            }
        }
#pragma unroll
    for (int ni = 0; ni < 2; ++ni)
        mx[ni] = fmaxf(mx[ni], __shfl_xor(mx[ni], 16, 32));

    if (hf == 0 && mbase < pool) {   // skip macro-tiles with no valid rows
#pragma unroll
        for (int ni = 0; ni < 2; ++ni) {
            const int col = n0 + 16 * ni + l15;
            atomicMax(&out_u[(size_t)b * DEMB + col], fkey(mx[ni] + b_out[col]));
        }
    }
}

// K5: decode order-preserving keys back to floats (in place).
__global__ void k5_decode(unsigned int* __restrict__ out_u,
                          float* __restrict__ out_f) {
    const int idx = blockIdx.x * blockDim.x + threadIdx.x;
    if (idx < BS * DEMB) out_f[idx] = funkey(out_u[idx]);
}

// ---------------------------------------------------------------------------
// Host side
// ---------------------------------------------------------------------------
extern "C" void kernel_launch(void* const* d_in, const int* in_sizes, int n_in,
                              void* d_out, int out_size, void* d_ws, size_t ws_size,
                              hipStream_t stream) {
    const float* features = (const float*)d_in[0];
    const int*   text     = (const int*)  d_in[1];
    // d_in[2] = atten: provably dead (row eos_pos fully overwritten with -1)
    const float* lin_w    = (const float*)d_in[3];
    const float* lin_b    = (const float*)d_in[4];
    const float* w1       = (const float*)d_in[5];
    // d_in[6] = b1: cancels exactly in train-mode BatchNorm
    const float* g1       = (const float*)d_in[7];
    const float* be1      = (const float*)d_in[8];
    const float* w2       = (const float*)d_in[9];
    const float* b2       = (const float*)d_in[10];
    // d_in[11]=g2, d_in[12]=be2: unused by reference

    char* w = (char*)d_ws;
    size_t o = 0;
    auto take = [&](size_t bytes) { void* p = w + o; o = (o + bytes + 255) & ~(size_t)255; return p; };
    unsigned short* Xbf   = (unsigned short*)take((size_t)MTOT * DIN * 2);   // 24 MB
    unsigned short* Hbf   = (unsigned short*)take((size_t)MTOT * DIN * 2);   // 24 MB
    unsigned short* LinWt = (unsigned short*)take((size_t)DEMB * DIN * 2);
    unsigned short* W1t   = (unsigned short*)take((size_t)512 * DIN * 2);
    unsigned short* W2t   = (unsigned short*)take((size_t)DEMB * DIN * 2);
    float* stats    = (float*)take(1024 * 4);
    float* scaleArr = (float*)take(512 * 4);
    float* shiftArr = (float*)take(512 * 4);
    float* b_out    = (float*)take(DEMB * 4);
    int*   pool     = (int*)  take(BS * 4);

    unsigned int* out_u = (unsigned int*)d_out;   // keys in-place, decoded by K5
    float*        out_f = (float*)d_out;

    k0_gather<<<BS, 256, 0, stream>>>(features, text, Xbf, pool);
    k1_prep<<<(DEMB * DIN) / 256, 256, 0, stream>>>(lin_w, w1, w2, lin_b, b2,
                                                    LinWt, W1t, W2t, b_out, stats, out_u);
    {
        dim3 g(512 / (8 * 32), MMACS);       // (2, 768)
        k2_gemm1_stats<<<g, 256, 0, stream>>>(Xbf, W1t, stats);
    }
    k3_finalize<<<2, 256, 0, stream>>>(stats, g1, be1, scaleArr, shiftArr);
    {
        dim3 g(512 / (8 * 32), MMACS);       // (2, 768)
        k2_gemm1_apply<<<g, 256, 0, stream>>>(Xbf, W1t, scaleArr, shiftArr, Hbf);
    }
    {
        dim3 g(DEMB / (8 * 32), MMACS);      // (4, 768)
        k4_gemm2_pool<<<g, 256, 0, stream>>>(Xbf, Hbf, LinWt, W2t, b_out, pool, out_u);
    }
    k5_decode<<<(BS * DEMB) / 256, 256, 0, stream>>>(out_u, out_f);
}
